// Sequence_33603824124322
// MI455X (gfx1250) — matrix operand order repaired and checked by reading, not compiled, not run on hardware
//
#include <hip/hip_runtime.h>
#include <hip/hip_bf16.h>

// ---------------- problem constants (from reference) ----------------
#define VARN   6912
#define FOURV  27648     // 4*VARN
#define BB     16
#define TT     8
#define TRR    4

typedef __bf16 v16bf __attribute__((ext_vector_type(16)));
typedef __bf16 v8bf  __attribute__((ext_vector_type(8)));
typedef float  v8f   __attribute__((ext_vector_type(8)));
typedef unsigned int u32x4 __attribute__((ext_vector_type(4)));
typedef int i32x4 __attribute__((ext_vector_type(4)));
typedef int i32x8 __attribute__((ext_vector_type(8)));

// therock-10.0 headers ship the 6-arg TDM builtin; ROCm 7.2 has the 5-arg one.
#if __has_include(<hip/amd_detail/amd_gfx1250_TDM.h>)
#define ATH_TDM_6ARG 1
#endif

// ---------------- helpers ----------------
__device__ __forceinline__ unsigned short f32_to_bf16_bits(float f) {
    union { float f; unsigned u; } v; v.f = f;
    unsigned u = v.u;
    u += 0x7FFFu + ((u >> 16) & 1u);   // round-to-nearest-even
    return (unsigned short)(u >> 16);
}
__device__ __forceinline__ __bf16 bits_to_bf16(unsigned short s) {
    union { unsigned short s; __bf16 b; } v; v.s = s;
    return v.b;
}
__device__ __forceinline__ __bf16 fbf(float f) { return bits_to_bf16(f32_to_bf16_bits(f)); }

__device__ __forceinline__ float sigmoid_f(float x) { return 1.0f / (1.0f + __expf(-x)); }

// ---------------- TDM: DMA one (16 rows x 32 K) weight tile into LDS ----------------
// D# packing per CDNA5 ISA ch.8 (group0 128b, group1 256b; groups 2/3 zero for 2D).
template <typename WT>
__device__ __forceinline__ void tdm_load_tile(const WT* W, int col0, int k0,
                                              int K, int N, unsigned lds_byte_off) {
    const unsigned long long ga =
        (unsigned long long)(const void*)(W + (size_t)col0 * (size_t)K + k0);
    u32x4 g0;
    g0[0] = 1u;                                            // count=1 valid user D#
    g0[1] = lds_byte_off;                                  // lds_addr
    g0[2] = (unsigned)(ga & 0xffffffffu);                  // global_addr[31:0]
    g0[3] = (unsigned)((ga >> 32) & 0x1ffffffu)            // global_addr[56:32]
          | (2u << 30);                                    // type = 2 ("image")
    const unsigned dsz = (sizeof(WT) == 2) ? 1u : 2u;      // 0=1B,1=2B,2=4B
    i32x8 g1;
    g1[0] = (int)(dsz << 16);                              // wg_mask=0, data_size
    g1[1] = (int)(((unsigned)K & 0xffffu) << 16);          // tensor_dim0[15:0] @ bits 63:48
    g1[2] = (int)((((unsigned)K >> 16) & 0xffffu)          // tensor_dim0[31:16]
          | (((unsigned)N & 0xffffu) << 16));              // tensor_dim1[15:0]
    g1[3] = (int)((((unsigned)N >> 16) & 0xffffu)          // tensor_dim1[31:16]
          | (32u << 16));                                  // tile_dim0 = 32
    g1[4] = 16;                                            // tile_dim1 = 16, tile_dim2 = 0
    g1[5] = K;                                             // tensor_dim0_stride[31:0]
    g1[6] = 0;                                             // stride0 hi, stride1 lo
    g1[7] = 0;
    i32x4 z4 = {0, 0, 0, 0};
#ifdef ATH_TDM_6ARG
    i32x8 z8 = {0, 0, 0, 0, 0, 0, 0, 0};
    __builtin_amdgcn_tensor_load_to_lds(g0, g1, z4, z4, z8, 0);
#else
    __builtin_amdgcn_tensor_load_to_lds(g0, g1, z4, z4, 0);
#endif
}

// ---------------- utility kernels ----------------
__global__ void zero_f32(float* p, int n) {
    int i = blockIdx.x * blockDim.x + threadIdx.x;
    int stride = gridDim.x * blockDim.x;
    for (; i < n; i += stride) p[i] = 0.0f;
}

__global__ void cvt_f32_to_bf16_v4(const float4* __restrict__ src,
                                   uint2* __restrict__ dst, int n4) {
    int i = blockIdx.x * blockDim.x + threadIdx.x;
    int stride = gridDim.x * blockDim.x;
    for (; i < n4; i += stride) {
        const float4 f = src[i];
        uint2 o;
        o.x = ((unsigned)f32_to_bf16_bits(f.y) << 16) | f32_to_bf16_bits(f.x);
        o.y = ((unsigned)f32_to_bf16_bits(f.w) << 16) | f32_to_bf16_bits(f.z);
        dst[i] = o;
    }
}

// ---------------- fragment builders ----------------
// A 16x32 bf16 fragment: lane m = lane&15, 16 contiguous K elements.
__device__ __forceinline__ v16bf load_a_frag(const unsigned short* aRun) {
    const v8bf r1 = *(const v8bf*)(aRun);
    const v8bf r2 = *(const v8bf*)(aRun + 8);
    return __builtin_shufflevector(r1, r2, 0, 1, 2, 3, 4, 5, 6, 7,
                                           8, 9, 10, 11, 12, 13, 14, 15);
}

// B 32x16 fragment: lane col = lane&15, 16 contiguous K elements from LDS tile.
template <typename WT>
__device__ __forceinline__ v16bf load_b_frag(const WT* bRow) {
    if constexpr (sizeof(WT) == 2) {
        return *(const v16bf*)bRow;                    // 32B -> 2x ds_load_b128
    } else {
        const float4* bp = (const float4*)bRow;
        const float4 b0 = bp[0], b1 = bp[1], b2 = bp[2], b3 = bp[3];
        v16bf b;
        b[0]  = fbf(b0.x); b[1]  = fbf(b0.y); b[2]  = fbf(b0.z); b[3]  = fbf(b0.w);
        b[4]  = fbf(b1.x); b[5]  = fbf(b1.y); b[6]  = fbf(b1.z); b[7]  = fbf(b1.w);
        b[8]  = fbf(b2.x); b[9]  = fbf(b2.y); b[10] = fbf(b2.z); b[11] = fbf(b2.w);
        b[12] = fbf(b3.x); b[13] = fbf(b3.y); b[14] = fbf(b3.z); b[15] = fbf(b3.w);
        return b;
    }
}

// One K-chunk: one B-tile load shared by MT independent accumulators (M-tiles).
template <int MT, typename WT>
__device__ __forceinline__ void wmma_chunk_mt(v8f (&acc)[MT],
                                              const unsigned short* A, int K,
                                              int rowFirst,  // mBase + mr
                                              int kpos,      // k0 + aOff
                                              const WT* bRow) {
    const v16bf b = load_b_frag<WT>(bRow);
#pragma unroll
    for (int mt = 0; mt < MT; ++mt) {
        const v16bf a =
            load_a_frag(A + (size_t)(rowFirst + mt * 16) * (size_t)K + kpos);
        acc[mt] = __builtin_amdgcn_wmma_f32_16x16x32_bf16(
            false, a, false, b, (short)0, acc[mt], false, false);
    }
}

// ---------------- WMMA GEMM, TDM-fed, 4-deep LDS weight-tile pipeline ----------------
// Out[m,n] = sum_k A[m,k] * W[n,k] (+ Cinit[m,n]) (+ bias[n])
// A is pre-converted bf16 (ushort bits). One wave per block (blockDim = 32)
// -> every TDM descriptor field is wave-uniform (SGPR-resident).
// MT M-tiles are folded into the wave so each weight byte crosses HBM exactly once.
template <int MT, typename WT>
__global__ __launch_bounds__(32)
void gemm_wmma_tdm(const unsigned short* __restrict__ A,
                   const WT* __restrict__ W,
                   const float* __restrict__ Cinit, int ldC,
                   const float* __restrict__ bias,
                   float* __restrict__ Out, int ldOut,
                   int N, int K) {
    __shared__ WT tile[4][16 * 32];        // 4-deep B-tile pipeline (16 rows x 32 K)

    const int lane  = threadIdx.x;
    const int nBase = blockIdx.x * 16;
    const int mBase = blockIdx.y * (MT * 16);
    const int half  = (lane >> 4) & 1;     // 0: lanes 0-15, 1: lanes 16-31
    const int mr    = lane & 15;
    const int aOff  = half * 16;           // A-fragment K offset (contiguous, matches B)
    const int bOff  = half * 16;           // B-fragment K offset within tile

    // Never true (grid sized so nBase < N); keeps the LDS object "written" so the
    // optimizer cannot fold the reads of the TDM-filled tile.
    if (nBase > N) tile[0][0] = (WT)0;

    const int rowFirst = mBase + mr;
    const int col      = nBase + mr;

    // Static shared layout: tile is the only LDS object -> starts at LDS offset 0.
    const unsigned bufBytes = (unsigned)(16 * 32 * sizeof(WT));

    // prologue: 3 DMAs in flight
    tdm_load_tile<WT>(W, nBase, 0,  K, N, 0u * bufBytes);
    tdm_load_tile<WT>(W, nBase, 32, K, N, 1u * bufBytes);
    tdm_load_tile<WT>(W, nBase, 64, K, N, 2u * bufBytes);

    v8f acc[MT];
#pragma unroll
    for (int mt = 0; mt < MT; ++mt) acc[mt] = (v8f){};

    int k0 = 0;
    for (; k0 + 96 < K; k0 += 32) {        // chunks 0 .. n-4
        const int buf = (k0 >> 5) & 3;
        tdm_load_tile<WT>(W, nBase, k0 + 96, K, N, (unsigned)((buf + 3) & 3) * bufBytes);
        __builtin_amdgcn_s_wait_tensorcnt(3);            // chunk k0 has landed
        asm volatile("" ::: "memory");                   // no LDS-read hoisting
        wmma_chunk_mt<MT, WT>(acc, A, K, rowFirst, k0 + aOff,
                              &tile[buf][mr * 32 + bOff]);
    }
    {   // drain: last three chunks, constant-immediate waits
        int buf = (k0 >> 5) & 3;
        __builtin_amdgcn_s_wait_tensorcnt(2);
        asm volatile("" ::: "memory");
        wmma_chunk_mt<MT, WT>(acc, A, K, rowFirst, k0 + aOff,
                              &tile[buf][mr * 32 + bOff]);
        k0 += 32; buf = (buf + 1) & 3;
        __builtin_amdgcn_s_wait_tensorcnt(1);
        asm volatile("" ::: "memory");
        wmma_chunk_mt<MT, WT>(acc, A, K, rowFirst, k0 + aOff,
                              &tile[buf][mr * 32 + bOff]);
        k0 += 32; buf = (buf + 1) & 3;
        __builtin_amdgcn_s_wait_tensorcnt(0);
        asm volatile("" ::: "memory");
        wmma_chunk_mt<MT, WT>(acc, A, K, rowFirst, k0 + aOff,
                              &tile[buf][mr * 32 + bOff]);
    }

    // C/D 16x16 f32 layout: VGPR r -> row r + half*8, col = lane&15.
    const float bv = bias ? bias[col] : 0.0f;
#pragma unroll
    for (int mt = 0; mt < MT; ++mt) {
#pragma unroll
        for (int r = 0; r < 8; ++r) {
            const int row = mBase + mt * 16 + r + half * 8;
            float v = acc[mt][r] + bv;
            if (Cinit) v += Cinit[(size_t)row * (size_t)ldC + col];
            Out[(size_t)row * (size_t)ldOut + col] = v;
        }
    }
}

// ---------------- encoder LSTM cell (gates -> h fp32 + bf16, c in place) ----------------
__global__ void lstm_cell_enc(const float* __restrict__ gates,
                              float* __restrict__ h, float* __restrict__ c,
                              unsigned short* __restrict__ h_bf) {
    int idx = blockIdx.x * blockDim.x + threadIdx.x;     // over B*V
    if (idx >= BB * VARN) return;
    const int b = idx / VARN, v = idx % VARN;
    const float* g = gates + (size_t)b * FOURV;
    const float ig = sigmoid_f(g[v]);
    const float fg = sigmoid_f(g[VARN + v]);
    const float gg = tanhf(g[2 * VARN + v]);
    const float og = sigmoid_f(g[3 * VARN + v]);
    const float cc = fg * c[idx] + ig * gg;
    const float hh = og * tanhf(cc);
    c[idx]    = cc;
    h[idx]    = hh;
    h_bf[idx] = f32_to_bf16_bits(hh);      // bf16 copy feeds the next WMMA GEMM
}

// ---------------- decoder cell (all TR steps independent; bf16 output) ----------------
__global__ void lstm_cell_dec(const float* __restrict__ hW,       // (B,4V)
                              const float* __restrict__ x_rev,    // (B,TR,1)
                              const float* __restrict__ W_ih_dec, // (4V,1)
                              const float* __restrict__ c_enc,    // (B,V)
                              unsigned short* __restrict__ h_d) { // (B,TR,V) bf16
    int idx = blockIdx.x * blockDim.x + threadIdx.x;     // over B*TR*V
    if (idx >= BB * TRR * VARN) return;
    const int v = idx % VARN;
    const int t = (idx / VARN) % TRR;
    const int b = idx / (VARN * TRR);
    const float xr = x_rev[b * TRR + t];
    const float* g = hW + (size_t)b * FOURV;
    const float ig = sigmoid_f(g[v]            + xr * W_ih_dec[v]);
    const float fg = sigmoid_f(g[VARN + v]     + xr * W_ih_dec[VARN + v]);
    const float gg = tanhf   (g[2 * VARN + v]  + xr * W_ih_dec[2 * VARN + v]);
    const float og = sigmoid_f(g[3 * VARN + v] + xr * W_ih_dec[3 * VARN + v]);
    const float cd = fg * c_enc[b * VARN + v] + ig * gg;
    h_d[idx] = f32_to_bf16_bits(og * tanhf(cd));
}

// ---------------- enc_out[b] = dot(h_enc[b,:], fc_enc_w) + fc_enc_b ----------------
__global__ void enc_fc(const float* __restrict__ h, const float* __restrict__ w,
                       const float* __restrict__ bias, float* __restrict__ out) {
    __shared__ float red[256];
    const int b = blockIdx.x;
    float s = 0.0f;
    for (int v = threadIdx.x; v < VARN; v += 256) s += h[(size_t)b * VARN + v] * w[v];
    red[threadIdx.x] = s;
    __syncthreads();
    for (int st = 128; st > 0; st >>= 1) {
        if ((int)threadIdx.x < st) red[threadIdx.x] += red[threadIdx.x + st];
        __syncthreads();
    }
    if (threadIdx.x == 0) out[b] = red[0] + bias[0];
}

// ---------------- dec_out = repeat(dec, TR-1, axis=1) ----------------
__global__ void repeat_out(const float* __restrict__ dec, float* __restrict__ out) {
    int idx = blockIdx.x * blockDim.x + threadIdx.x;     // over B*12*V
    const int total = BB * (TRR * (TRR - 1)) * VARN;
    if (idx >= total) return;
    const int v = idx % VARN;
    const int j = (idx / VARN) % (TRR * (TRR - 1));
    const int b = idx / (VARN * TRR * (TRR - 1));
    out[idx] = dec[((size_t)b * TRR + (j / (TRR - 1))) * VARN + v];
}

// ---------------- host launcher ----------------
extern "C" void kernel_launch(void* const* d_in, const int* in_sizes, int n_in,
                              void* d_out, int out_size, void* d_ws, size_t ws_size,
                              hipStream_t stream) {
    const float* x        = (const float*)d_in[0];
    const float* x_rev    = (const float*)d_in[1];
    const float* W_ih_enc = (const float*)d_in[2];
    const float* W_hh_enc = (const float*)d_in[3];
    const float* b_enc    = (const float*)d_in[4];
    const float* fc_enc_w = (const float*)d_in[5];
    const float* fc_enc_b = (const float*)d_in[6];
    const float* W_ih_dec = (const float*)d_in[7];
    const float* W_hh_dec = (const float*)d_in[8];
    const float* b_dec    = (const float*)d_in[9];
    const float* fc_dec_w = (const float*)d_in[10];
    const float* fc_dec_b = (const float*)d_in[11];
    float* out = (float*)d_out;
    (void)in_sizes; (void)n_in; (void)out_size; (void)ws_size;

    // workspace carve-up
    char* ws = (char*)d_ws;
    size_t off = 0;
    auto take = [&](size_t bytes) -> char* {
        char* p = ws + off;
        off = (off + bytes + 255) & ~(size_t)255;
        return p;
    };
    unsigned short* whh_bf = (unsigned short*)take((size_t)FOURV * VARN * 2); // 382 MB
    unsigned short* x_bf   = (unsigned short*)take((size_t)BB * TT * VARN * 2);
    unsigned short* h_bf   = (unsigned short*)take((size_t)BB * VARN * 2);
    unsigned short* hd_bf  = (unsigned short*)take((size_t)BB * TRR * VARN * 2);
    float* xg    = (float*)take((size_t)BB * TT * FOURV * 4);   // (B*T, 4V) b-major
    float* gates = (float*)take((size_t)BB * FOURV * 4);        // (B, 4V) scratch
    float* hc    = (float*)take((size_t)2 * BB * VARN * 4);     // h then c
    float* h     = hc;
    float* c     = hc + (size_t)BB * VARN;
    float* decb  = (float*)take((size_t)BB * TRR * VARN * 4);

    // 0) h0 = c0 = 0 (fp32 and bf16 copies)
    zero_f32<<<(2 * BB * VARN + 255) / 256, 256, 0, stream>>>(hc, 2 * BB * VARN);
    zero_f32<<<(BB * VARN / 2 + 255) / 256, 256, 0, stream>>>(
        (float*)h_bf, BB * VARN / 2);

    // 1) pre-convert: 8x-reused recurrence weight (saves ~1.9 GB HBM) and x
    cvt_f32_to_bf16_v4<<<4096, 256, 0, stream>>>(
        (const float4*)W_hh_enc, (uint2*)whh_bf, (FOURV * VARN) / 4);
    cvt_f32_to_bf16_v4<<<(BB * TT * VARN / 4 + 255) / 256, 256, 0, stream>>>(
        (const float4*)x, (uint2*)x_bf, (BB * TT * VARN) / 4);

    // 2) xg = x @ W_ih_enc^T + b_enc   (M=128 folded into the wave: MT=8)
    {
        dim3 grid(FOURV / 16, 1);
        gemm_wmma_tdm<8><<<grid, 32, 0, stream>>>(x_bf, W_ih_enc,
                                                  (const float*)nullptr, 0,
                                                  b_enc, xg, FOURV, FOURV, VARN);
    }

    // 3) encoder recurrence: gates = xg[:,t,:] + h @ W_hh_enc^T ; cell update
    {
        dim3 grid(FOURV / 16, 1);
        for (int t = 0; t < TT; ++t) {
            gemm_wmma_tdm<1><<<grid, 32, 0, stream>>>(h_bf, whh_bf,
                                                      xg + (size_t)t * FOURV, TT * FOURV,
                                                      (const float*)nullptr,
                                                      gates, FOURV, FOURV, VARN);
            lstm_cell_enc<<<(BB * VARN + 255) / 256, 256, 0, stream>>>(
                gates, h, c, h_bf);
        }
    }

    // 4) enc_out = h_enc @ fc_enc_w^T + fc_enc_b   -> out[0..15]
    enc_fc<<<BB, 256, 0, stream>>>(h, fc_enc_w, fc_enc_b, out);

    // 5) hW = h_enc @ W_hh_dec^T + b_dec  (reuse gates buffer)
    {
        dim3 grid(FOURV / 16, 1);
        gemm_wmma_tdm<1><<<grid, 32, 0, stream>>>(h_bf, W_hh_dec,
                                                  (const float*)nullptr, 0,
                                                  b_dec, gates, FOURV, FOURV, VARN);
    }

    // 6) decoder cell (TR independent steps) -> h_d (bf16)
    lstm_cell_dec<<<(BB * TRR * VARN + 255) / 256, 256, 0, stream>>>(
        gates, x_rev, W_ih_dec, c, hd_bf);

    // 7) dec = h_d @ fc_dec_w^T + fc_dec_b  (M=64 folded into the wave: MT=4)
    {
        dim3 grid(VARN / 16, 1);
        gemm_wmma_tdm<4><<<grid, 32, 0, stream>>>(hd_bf, fc_dec_w,
                                                  (const float*)nullptr, 0,
                                                  fc_dec_b, decb, VARN, VARN, VARN);
    }

    // 8) dec_out = repeat(dec, TR-1, axis=1) -> out[16 ...]
    repeat_out<<<(BB * (TRR * (TRR - 1)) * VARN + 255) / 256, 256, 0, stream>>>(
        decb, out + BB);
}
